// GAT_14078902796504
// MI455X (gfx1250) — matrix operand order repaired and checked by reading, not compiled, not run on hardware
//
#include <hip/hip_runtime.h>
#include <hip/hip_bf16.h>

typedef __attribute__((ext_vector_type(16))) __bf16 v16bf;
typedef __attribute__((ext_vector_type(8)))  float  v8f;

#define NN     4096
#define NFEATC 256
#define NHID   16
#define NHEADS 8
#define LALPHA 0.2f
#define NTILES (NN / 32)

// ---------------------------------------------------------------------------
// Kernel 1: per-head feature GEMM  Wh[h] = X @ W[h]   ([N,F] x [F,16] -> [N,16])
// One wave per 16-row output tile. bf16 WMMA 16x16x32, f32 accumulate.
// ---------------------------------------------------------------------------
__global__ void gat_xw(const float* __restrict__ X, const float* __restrict__ W,
                       __bf16* __restrict__ Wh, __bf16* __restrict__ WhT, int F) {
  const int h    = blockIdx.x >> 8;
  const int i0   = (blockIdx.x & 255) << 4;
  const int lane = threadIdx.x;
  const int row  = lane & 15;
  const int hi   = lane >> 4;
  const int koffA = hi << 3;
  const int koffB = hi << 4;
  const float* Wp = W + (size_t)h * F * NHID;

  v8f acc = {};
  for (int k0 = 0; k0 < F; k0 += 32) {
    const float* ab = X + (size_t)(i0 + row) * F + k0 + koffA;
    float4 x0 = *(const float4*)(ab + 0);
    float4 x1 = *(const float4*)(ab + 4);
    float4 x2 = *(const float4*)(ab + 16);
    float4 x3 = *(const float4*)(ab + 20);
    float av[16] = {x0.x, x0.y, x0.z, x0.w, x1.x, x1.y, x1.z, x1.w,
                    x2.x, x2.y, x2.z, x2.w, x3.x, x3.y, x3.z, x3.w};
    v16bf a;
#pragma unroll
    for (int t = 0; t < 16; ++t) a[t] = (__bf16)av[t];

    v16bf b;
#pragma unroll
    for (int t = 0; t < 16; ++t)
      b[t] = (__bf16)Wp[(size_t)(k0 + koffA + (t & 7) + ((t >> 3) << 4)) * NHID + row];

    acc = __builtin_amdgcn_wmma_f32_16x16x32_bf16(false, a, false, b,
                                                  (short)0, acc, false, false);
  }

  const int d = row;
#pragma unroll
  for (int v = 0; v < 8; ++v) {
    int M = i0 + v + (hi << 3);
    __bf16 w = (__bf16)acc[v];
    Wh [((size_t)h * NN + M) * NHID + d] = w;
    WhT[((size_t)h * NHID + d) * NN + M] = w;
  }
}

// ---------------------------------------------------------------------------
// Kernel 2: src[h,n] = Wh[h,n,:].a_src[h]   dst[h,n] = Wh[h,n,:].a_dst[h]
// ---------------------------------------------------------------------------
__global__ void gat_srcdst(const __bf16* __restrict__ Wh,
                           const float* __restrict__ a_s, const float* __restrict__ a_d,
                           float* __restrict__ src, float* __restrict__ dst, int total) {
  int tid = blockIdx.x * blockDim.x + threadIdx.x;
  if (tid >= total) return;
  int h = tid / NN;
  const __bf16* wrow = Wh + (size_t)tid * NHID;
  const float* as = a_s + h * NHID;
  const float* ad = a_d + h * NHID;
  float s = 0.f, dv = 0.f;
#pragma unroll
  for (int d = 0; d < NHID; ++d) {
    float w = (float)wrow[d];
    s  += w * as[d];
    dv += w * ad[d];
  }
  src[tid] = s;
  dst[tid] = dv;
}

// ---------------------------------------------------------------------------
// Kernel 3a (layer 1): all-heads-fused attention, 8 waves per block (1 head
// per wave), adjacency tile staged ONCE per row-block through LDS via
// double-buffered GLOBAL_LOAD_ASYNC_TO_LDS_B128 (ASYNCcnt) issued by wave 0:
//   issue load(t+1) -> compute tile t from LDS + bf16 WMMA -> s_wait_asynccnt
//   -> workgroup barrier.  Cuts layer-1 adj read traffic 8x.
// ---------------------------------------------------------------------------
__global__ void gat_attn_fused(const __bf16* __restrict__ WhT,
                               const float* __restrict__ src, const float* __restrict__ dst,
                               const float* __restrict__ adj,
                               float* __restrict__ out) {
  __shared__ float tile[2][16 * 32];     // 2 x 2KB adj tile double buffer
  __shared__ float lred[NHEADS][16];

  const int i0   = blockIdx.x << 4;
  const int tid  = threadIdx.x;
  const int w    = tid >> 5;             // wave index == head
  const int lane = tid & 31;
  const int row  = lane & 15;
  const int hi   = lane >> 4;
  const int d    = row;
  const int koffA = hi << 3;
  const int h    = w;

  // loader lane mapping (wave 0): 4 async b128 ops move the 16x32 f32 tile
  const int lr = lane >> 3;              // row-within-group 0..3
  const int lc = (lane & 7) << 2;        // col offset 0,4,...,28

  const float  s_i  = src[h * NN + i0 + row];
  const float* dstp = dst + h * NN + koffA;
  const __bf16* bcol = WhT + ((size_t)h * NHID + d) * NN + (hi << 3);

  v8f acc = {};
  float lsum = 0.f;

  if (w == 0) {                          // preload tile 0 into buffer 0
#pragma unroll
    for (int q = 0; q < 4; ++q) {
      int r = (q << 2) + lr;
      const float* g = adj + (size_t)(i0 + r) * NN + lc;
      unsigned lds = (unsigned)(uintptr_t)&tile[0][r * 32 + lc];
      asm volatile("global_load_async_to_lds_b128 %0, %1, off"
                   :: "v"(lds), "v"(g) : "memory");
    }
    asm volatile("s_wait_asynccnt 0" ::: "memory");
  }
  __syncthreads();

  for (int jt = 0; jt < NTILES; ++jt) {
    const int p = jt & 1;
    if (w == 0 && jt + 1 < NTILES) {     // overlap: stage tile jt+1
      const int j1 = (jt + 1) << 5;
#pragma unroll
      for (int q = 0; q < 4; ++q) {
        int r = (q << 2) + lr;
        const float* g = adj + (size_t)(i0 + r) * NN + j1 + lc;
        unsigned lds = (unsigned)(uintptr_t)&tile[p ^ 1][r * 32 + lc];
        asm volatile("global_load_async_to_lds_b128 %0, %1, off"
                     :: "v"(lds), "v"(g) : "memory");
      }
    }

    const int j0 = jt << 5;
    const float* tp = &tile[p][row * 32 + koffA];   // ds_load_b128 reads
    float4 m0 = *(const float4*)(tp + 0);
    float4 m1 = *(const float4*)(tp + 4);
    float4 m2 = *(const float4*)(tp + 16);
    float4 m3 = *(const float4*)(tp + 20);
    const float* dp = dstp + j0;
    float4 d0 = *(const float4*)(dp + 0);
    float4 d1 = *(const float4*)(dp + 4);
    float4 d2 = *(const float4*)(dp + 16);
    float4 d3 = *(const float4*)(dp + 20);

    float mv[16] = {m0.x, m0.y, m0.z, m0.w, m1.x, m1.y, m1.z, m1.w,
                    m2.x, m2.y, m2.z, m2.w, m3.x, m3.y, m3.z, m3.w};
    float dv[16] = {d0.x, d0.y, d0.z, d0.w, d1.x, d1.y, d1.z, d1.w,
                    d2.x, d2.y, d2.z, d2.w, d3.x, d3.y, d3.z, d3.w};

    v16bf a;
#pragma unroll
    for (int t = 0; t < 16; ++t) {
      float z = s_i + dv[t];
      z = z > 0.f ? z : LALPHA * z;                 // leaky_relu(0.2)
      float pv = mv[t] > 0.f ? __expf(z) : 0.f;
      lsum += pv;
      a[t] = (__bf16)pv;
    }

    typedef __attribute__((ext_vector_type(8))) __bf16 v8bf_; v8bf_ b0 = *(const v8bf_*)(bcol + j0), b1 = *(const v8bf_*)(bcol + j0 + 16); v16bf b = __builtin_shufflevector(b0, b1, 0, 1, 2, 3, 4, 5, 6, 7, 8, 9, 10, 11, 12, 13, 14, 15);
    acc = __builtin_amdgcn_wmma_f32_16x16x32_bf16(false, a, false, b,
                                                  (short)0, acc, false, false);

    if (w == 0) asm volatile("s_wait_asynccnt 0" ::: "memory");
    __syncthreads();
  }

  // per-wave row-sum reduction (row r partials in lanes r and r+16)
  if (hi == 0) lred[w][row] = lsum;
  __syncthreads();
  if (hi == 1) lred[w][row] += lsum;
  __syncthreads();

#pragma unroll
  for (int v = 0; v < 8; ++v) {
    int M = v + (hi << 3);
    float r = acc[v] / lred[w][M];                  // softmax normalize
    r = r > 0.f ? r : (__expf(r) - 1.f);            // inner ELU (concat layer)
    out[(size_t)(i0 + M) * (NHEADS * NHID) + h * NHID + d] = r;
  }
}

// ---------------------------------------------------------------------------
// Kernel 3b (layer 2): single-head streaming attention, one wave per 16-row
// block (identical math; adj read once, L2-resident).
// ---------------------------------------------------------------------------
__global__ void gat_attn(const __bf16* __restrict__ WhT,
                         const float* __restrict__ src, const float* __restrict__ dst,
                         const float* __restrict__ adj,
                         float* __restrict__ out, int ostride) {
  const int h    = blockIdx.x >> 8;
  const int i0   = (blockIdx.x & 255) << 4;
  const int lane = threadIdx.x;
  const int row  = lane & 15;
  const int hi   = lane >> 4;
  const int d    = row;
  const int koffA = hi << 3;

  const float  s_i  = src[h * NN + i0 + row];
  const float* dstp = dst + h * NN + koffA;
  const float* arow = adj + (size_t)(i0 + row) * NN + koffA;
  const __bf16* bcol = WhT + ((size_t)h * NHID + d) * NN + (hi << 3);

  v8f acc = {};
  float lsum = 0.f;

  for (int j0 = 0; j0 < NN; j0 += 32) {
    float4 m0 = *(const float4*)(arow + j0 + 0);
    float4 m1 = *(const float4*)(arow + j0 + 4);
    float4 m2 = *(const float4*)(arow + j0 + 16);
    float4 m3 = *(const float4*)(arow + j0 + 20);
    float4 d0 = *(const float4*)(dstp + j0 + 0);
    float4 d1 = *(const float4*)(dstp + j0 + 4);
    float4 d2 = *(const float4*)(dstp + j0 + 16);
    float4 d3 = *(const float4*)(dstp + j0 + 20);
    __builtin_prefetch(arow + j0 + 32, 0, 0);

    float mv[16] = {m0.x, m0.y, m0.z, m0.w, m1.x, m1.y, m1.z, m1.w,
                    m2.x, m2.y, m2.z, m2.w, m3.x, m3.y, m3.z, m3.w};
    float dv[16] = {d0.x, d0.y, d0.z, d0.w, d1.x, d1.y, d1.z, d1.w,
                    d2.x, d2.y, d2.z, d2.w, d3.x, d3.y, d3.z, d3.w};

    v16bf a;
#pragma unroll
    for (int t = 0; t < 16; ++t) {
      float z = s_i + dv[t];
      z = z > 0.f ? z : LALPHA * z;
      float pv = mv[t] > 0.f ? __expf(z) : 0.f;
      lsum += pv;
      a[t] = (__bf16)pv;
    }

    typedef __attribute__((ext_vector_type(8))) __bf16 v8bf_; v8bf_ b0 = *(const v8bf_*)(bcol + j0), b1 = *(const v8bf_*)(bcol + j0 + 16); v16bf b = __builtin_shufflevector(b0, b1, 0, 1, 2, 3, 4, 5, 6, 7, 8, 9, 10, 11, 12, 13, 14, 15);
    acc = __builtin_amdgcn_wmma_f32_16x16x32_bf16(false, a, false, b,
                                                  (short)0, acc, false, false);
  }

  __shared__ float lred[16];
  if (hi == 0) lred[row] = lsum;
  __syncthreads();
  if (hi == 1) lred[row] += lsum;
  __syncthreads();

#pragma unroll
  for (int v = 0; v < 8; ++v) {
    int M = v + (hi << 3);
    float r = acc[v] / lred[M];
    r = r > 0.f ? r : (__expf(r) - 1.f);            // outer ELU
    out[(size_t)(i0 + M) * ostride + h * NHID + d] = r;
  }
}

// ---------------------------------------------------------------------------
extern "C" void kernel_launch(void* const* d_in, const int* in_sizes, int n_in,
                              void* d_out, int out_size, void* d_ws, size_t ws_size,
                              hipStream_t stream) {
  const float* x         = (const float*)d_in[0];  // [4096,256]
  const float* adj       = (const float*)d_in[1];  // [4096,4096]
  const float* W         = (const float*)d_in[2];  // [8,256,16]
  const float* a_src     = (const float*)d_in[3];  // [8,16]
  const float* a_dst     = (const float*)d_in[4];  // [8,16]
  const float* W_out     = (const float*)d_in[5];  // [128,16]
  const float* a_out_src = (const float*)d_in[6];  // [16]
  const float* a_out_dst = (const float*)d_in[7];  // [16]
  float* out = (float*)d_out;                      // [4096,16]

  char* ws = (char*)d_ws;
  size_t off = 0;
  __bf16* Wh1  = (__bf16*)(ws + off); off += (size_t)NHEADS * NN * NHID * 2;
  __bf16* WhT1 = (__bf16*)(ws + off); off += (size_t)NHEADS * NN * NHID * 2;
  float*  src1 = (float*) (ws + off); off += (size_t)NHEADS * NN * 4;
  float*  dst1 = (float*) (ws + off); off += (size_t)NHEADS * NN * 4;
  float*  h2   = (float*) (ws + off); off += (size_t)NN * NHEADS * NHID * 4;
  __bf16* Wh2  = (__bf16*)(ws + off); off += (size_t)NN * NHID * 2;
  __bf16* WhT2 = (__bf16*)(ws + off); off += (size_t)NN * NHID * 2;
  float*  src2 = (float*) (ws + off); off += (size_t)NN * 4;
  float*  dst2 = (float*) (ws + off); off += (size_t)NN * 4;

  // ---- layer 1 (8 heads fused, concat, inner ELU) ----
  gat_xw        <<<dim3(NHEADS * 256), dim3(32),  0, stream>>>(x, W, Wh1, WhT1, NFEATC);
  gat_srcdst    <<<dim3(NHEADS * NN / 256), dim3(256), 0, stream>>>(Wh1, a_src, a_dst,
                                                                    src1, dst1, NHEADS * NN);
  gat_attn_fused<<<dim3(256), dim3(NHEADS * 32), 0, stream>>>(WhT1, src1, dst1, adj, h2);

  // ---- layer 2 (1 head, outer ELU) ----
  gat_xw        <<<dim3(256), dim3(32),  0, stream>>>(h2, W_out, Wh2, WhT2, NHEADS * NHID);
  gat_srcdst    <<<dim3(NN / 256), dim3(256), 0, stream>>>(Wh2, a_out_src, a_out_dst,
                                                           src2, dst2, NN);
  gat_attn      <<<dim3(256), dim3(32),  0, stream>>>(WhT2, src2, dst2, adj, out, NHID);
}
